// SparseMoEFFN_49795850830456
// MI455X (gfx1250) — matrix operand order repaired and checked by reading, not compiled, not run on hardware
//
#include <hip/hip_runtime.h>
#include <hip/hip_bf16.h>

typedef __attribute__((ext_vector_type(16))) __bf16 v16bf;
typedef __attribute__((ext_vector_type(8)))  float  v8f;

#define Dm 768
#define EDm 192
#define Em 8
#define Tm 32768
#define CAPm (2*Tm + Em*64)   // 66048 assignment slots, expert segments padded to 64
#define LN_EPS 1e-5f

union Frag { uint4 u4[2]; v16bf v; };

// ---------------- transpose fp32 [R][C] -> bf16 [C][R], per-matrix z ----------------
__global__ __launch_bounds__(256) void k_transpose(const float* __restrict__ src,
                                                   __hip_bfloat16* __restrict__ dst,
                                                   int R, int C) {
  __shared__ float tile[32][33];
  const float* s = src + (size_t)blockIdx.z * R * C;
  __hip_bfloat16* d = dst + (size_t)blockIdx.z * R * C;
  int c0 = blockIdx.x * 32, r0 = blockIdx.y * 32;
  int tx = threadIdx.x & 31, ty = threadIdx.x >> 5;
#pragma unroll
  for (int i = 0; i < 32; i += 8)
    tile[ty + i][tx] = s[(size_t)(r0 + ty + i) * C + (c0 + tx)];
  __syncthreads();
#pragma unroll
  for (int i = 0; i < 32; i += 8)
    d[(size_t)(c0 + ty + i) * R + (r0 + tx)] = __float2bfloat16(tile[tx][ty + i]);
}

// ---------------- zero counters, fill perm/gate padding ----------------
__global__ void k_init(int* __restrict__ perm, float* __restrict__ gatep,
                       int* __restrict__ counts, int* __restrict__ cursor) {
  int i = blockIdx.x * 256 + threadIdx.x;
  if (i < CAPm) { perm[i] = 0; gatep[i] = 0.f; }
  if (i < Em)  { counts[i] = 0; cursor[i] = 0; }
}

// ---------------- router: wave per token ----------------
__global__ __launch_bounds__(256) void k_router(const __hip_bfloat16* __restrict__ tokA,
                                                const float* __restrict__ Wr,
                                                int* __restrict__ eid, float* __restrict__ gate,
                                                int* __restrict__ counts) {
  __shared__ float wr[Em * Dm];
  for (int i = threadIdx.x; i < Em * Dm; i += 256) wr[i] = Wr[i];
  __syncthreads();
  int wv = threadIdx.x >> 5, lane = threadIdx.x & 31;
  int t = blockIdx.x * 8 + wv;
  float acc[Em];
#pragma unroll
  for (int e = 0; e < Em; ++e) acc[e] = 0.f;
  const __hip_bfloat16* row = tokA + (size_t)t * Dm;
  for (int d = lane; d < Dm; d += 32) {
    float v = __bfloat162float(row[d]);
#pragma unroll
    for (int e = 0; e < Em; ++e) acc[e] += v * wr[e * Dm + d];
  }
#pragma unroll
  for (int e = 0; e < Em; ++e)
#pragma unroll
    for (int off = 16; off; off >>= 1) acc[e] += __shfl_xor(acc[e], off, 32);
  if (lane == 0) {
    int i0 = 0; float v0 = acc[0];
    for (int e = 1; e < Em; ++e) if (acc[e] > v0) { v0 = acc[e]; i0 = e; }
    int i1 = -1; float v1 = -3.4e38f;
    for (int e = 0; e < Em; ++e) if (e != i0 && acc[e] > v1) { v1 = acc[e]; i1 = e; }
    float w1 = __expf(v1 - v0);          // <= 1
    float inv = 1.f / (1.f + w1);
    eid[2 * t] = i0; eid[2 * t + 1] = i1;
    gate[2 * t] = inv; gate[2 * t + 1] = w1 * inv;
    atomicAdd(&counts[i0], 1); atomicAdd(&counts[i1], 1);
  }
}

// ---------------- tiny exclusive scan with 64-row padding ----------------
__global__ void k_scan(const int* __restrict__ counts, int* __restrict__ offs) {
  if (threadIdx.x == 0) {
    int acc = 0;
    for (int e = 0; e < Em; ++e) { offs[e] = acc; acc += (counts[e] + 63) & ~63; }
    offs[Em] = acc;
  }
}

// ---------------- scatter assignments into expert segments ----------------
__global__ __launch_bounds__(256) void k_scatter(const int* __restrict__ eid,
                                                 const float* __restrict__ gate,
                                                 const int* __restrict__ offs,
                                                 int* __restrict__ cursor,
                                                 int* __restrict__ perm,
                                                 float* __restrict__ gatep,
                                                 int* __restrict__ slot_of) {
  int a = blockIdx.x * 256 + threadIdx.x;
  if (a >= 2 * Tm) return;
  int e = eid[a];
  int pos = atomicAdd(&cursor[e], 1);
  int slot = offs[e] + pos;
  perm[slot] = a >> 1;
  gatep[slot] = gate[a];
  slot_of[a] = slot;
}

// ---------------- GEMM1: H = silu(A@W1) * (A@W3), 64-row gathered tiles ----------------
// waves 0-3 compute the W1 half, waves 4-7 the W3 half (same m/n mapping);
// SiLU epilogue exchanges the W1 results through LDS (aliased onto the A buffer).
__global__ __launch_bounds__(256) void k_gemm1(const __hip_bfloat16* __restrict__ tokA,
                                               const __hip_bfloat16* __restrict__ W1t,
                                               const __hip_bfloat16* __restrict__ W3t,
                                               const int* __restrict__ offs,
                                               const int* __restrict__ perm,
                                               __hip_bfloat16* __restrict__ H) {
  __shared__ uint4 ldsRaw[(64 * Dm * 2) / 16];    // 96 KB: A tile, later fp32 exchange
  __shared__ int ptok[64];
  __hip_bfloat16* ldsA = (__hip_bfloat16*)ldsRaw;
  float* ldsF = (float*)ldsRaw;                   // 64*192 fp32 = 48 KB (aliased)
  int base = blockIdx.x * 64;
  if (base >= offs[Em]) return;
  int e = 0;
  while (e < Em - 1 && base >= offs[e + 1]) ++e;
  int tid = threadIdx.x;
  if (tid < 64) ptok[tid] = perm[base + tid];
  __syncthreads();
  for (int p = tid; p < 64 * (Dm / 8); p += 256) {
    int r = p / (Dm / 8), ch = p % (Dm / 8);
    ((uint4*)ldsA)[p] = ((const uint4*)(tokA + (size_t)ptok[r] * Dm))[ch];
  }
  __syncthreads();
  int wv = tid >> 5, lane = tid & 31, lo = lane & 15, hi = lane >> 4;
  int mat = wv >> 2;                   // 0 -> W1 branch, 1 -> W3 branch
  int nw  = wv & 3;                    // n-group: 3 n-tiles each, 4*3*16 = 192 = EDm
  const __hip_bfloat16* B = (mat ? W3t : W1t) + (size_t)e * EDm * Dm;  // [n][d]
  v8f z = {0.f,0.f,0.f,0.f,0.f,0.f,0.f,0.f};
  v8f acc[4][3];
#pragma unroll
  for (int mi = 0; mi < 4; ++mi)
#pragma unroll
    for (int j = 0; j < 3; ++j) acc[mi][j] = z;
  for (int kk = 0; kk < Dm / 32; ++kk) {
    Frag a[4];
#pragma unroll
    for (int mi = 0; mi < 4; ++mi) {
      const __hip_bfloat16* ap = ldsA + (mi * 16 + lo) * Dm + kk * 32 + hi * 8;
      a[mi].u4[0] = *(const uint4*)ap;
      a[mi].u4[1] = *(const uint4*)(ap + 16);
    }
    int d0 = kk * 32 + hi * 8;
#pragma unroll
    for (int j = 0; j < 3; ++j) {
      int n = (nw * 3 + j) * 16 + lo;
      Frag b;
      const uint4* bp = (const uint4*)(B + (size_t)n * Dm + d0);
      b.u4[0] = bp[0]; b.u4[1] = bp[2];
#pragma unroll
      for (int mi = 0; mi < 4; ++mi)
        acc[mi][j] = __builtin_amdgcn_wmma_f32_16x16x32_bf16(false, a[mi].v, false, b.v,
                                                             (short)0, acc[mi][j], false, false);
    }
  }
  __syncthreads();          // done with ldsA; reuse as fp32 exchange buffer
  if (mat == 0) {
#pragma unroll
    for (int mi = 0; mi < 4; ++mi)
#pragma unroll
      for (int j = 0; j < 3; ++j) {
        int n = (nw * 3 + j) * 16 + lo;
#pragma unroll
        for (int i = 0; i < 8; ++i) {
          int m = mi * 16 + i + hi * 8;
          ldsF[m * EDm + n] = acc[mi][j][i];
        }
      }
  }
  __syncthreads();
  if (mat == 1) {
#pragma unroll
    for (int mi = 0; mi < 4; ++mi)
#pragma unroll
      for (int j = 0; j < 3; ++j) {
        int n = (nw * 3 + j) * 16 + lo;
#pragma unroll
        for (int i = 0; i < 8; ++i) {
          int m = mi * 16 + i + hi * 8;
          float g1 = ldsF[m * EDm + n];
          float s = g1 / (1.f + __expf(-g1));
          H[(size_t)(base + m) * EDm + n] = __float2bfloat16(s * acc[mi][j][i]);
        }
      }
  }
}

// ---------------- GEMM2: Yrow = gate * (H @ W2), 64-row x 384-col tiles ----------------
__global__ __launch_bounds__(256) void k_gemm2(const __hip_bfloat16* __restrict__ H,
                                               const __hip_bfloat16* __restrict__ W2t,
                                               const int* __restrict__ offs,
                                               const float* __restrict__ gatep,
                                               __hip_bfloat16* __restrict__ Yr) {
  __shared__ __hip_bfloat16 ldsH[64 * EDm];   // 24 KB
  __shared__ float ldsG[64];
  int base = blockIdx.x * 64;
  if (base >= offs[Em]) return;
  int band = blockIdx.y;                       // 0 or 1: n in [band*384, band*384+384)
  int e = 0;
  while (e < Em - 1 && base >= offs[e + 1]) ++e;
  int tid = threadIdx.x;
  if (tid < 64) ldsG[tid] = gatep[base + tid];
  for (int p = tid; p < 64 * (EDm / 8); p += 256) {
    int r = p / (EDm / 8), ch = p % (EDm / 8);
    ((uint4*)ldsH)[p] = ((const uint4*)(H + (size_t)(base + r) * EDm))[ch];
  }
  __syncthreads();
  int wv = tid >> 5, lane = tid & 31, lo = lane & 15, hi = lane >> 4;
  v8f z = {0.f,0.f,0.f,0.f,0.f,0.f,0.f,0.f};
  v8f acc[4][3];
#pragma unroll
  for (int mi = 0; mi < 4; ++mi)
#pragma unroll
    for (int j = 0; j < 3; ++j) acc[mi][j] = z;
  const __hip_bfloat16* B = W2t + (size_t)e * Dm * EDm;   // [n][k], k contiguous
  for (int kk = 0; kk < EDm / 32; ++kk) {
    Frag a[4];
#pragma unroll
    for (int mi = 0; mi < 4; ++mi) {
      const __hip_bfloat16* ap = ldsH + (mi * 16 + lo) * EDm + kk * 32 + hi * 8;
      a[mi].u4[0] = *(const uint4*)ap;
      a[mi].u4[1] = *(const uint4*)(ap + 16);
    }
    int k0 = kk * 32 + hi * 8;
#pragma unroll
    for (int j = 0; j < 3; ++j) {
      int n = (band * 24 + wv * 3 + j) * 16 + lo;
      Frag b;
      const uint4* bp = (const uint4*)(B + (size_t)n * EDm + k0);
      b.u4[0] = bp[0]; b.u4[1] = bp[2];
#pragma unroll
      for (int mi = 0; mi < 4; ++mi)
        acc[mi][j] = __builtin_amdgcn_wmma_f32_16x16x32_bf16(false, a[mi].v, false, b.v,
                                                             (short)0, acc[mi][j], false, false);
    }
  }
#pragma unroll
  for (int mi = 0; mi < 4; ++mi)
#pragma unroll
    for (int j = 0; j < 3; ++j) {
      int n = (band * 24 + wv * 3 + j) * 16 + lo;
#pragma unroll
      for (int i = 0; i < 8; ++i) {
        int m = mi * 16 + i + hi * 8;
        Yr[(size_t)(base + m) * Dm + n] = __float2bfloat16(acc[mi][j][i] * ldsG[m]);
      }
    }
}

// ---------------- combine + layernorm + residual, block per (b,h) ----------------
__global__ __launch_bounds__(256) void k_final(const __hip_bfloat16* __restrict__ Yr,
                                               const int* __restrict__ slot_of,
                                               const float* __restrict__ x,
                                               const float* __restrict__ gamma,
                                               const float* __restrict__ beta,
                                               float* __restrict__ out) {
  __shared__ float yb[32 * 770];             // pitch 770 avoids bank conflicts
  __shared__ float muS[32], rsS[32];
  int b = blockIdx.x >> 5, h = blockIdx.x & 31;
  int tid = threadIdx.x, wv = tid >> 5, lane = tid & 31;
#pragma unroll
  for (int q = 0; q < 4; ++q) {
    int w = wv + q * 8;
    int t = b * 1024 + h * 32 + w;
    const __hip_bfloat16* r0 = Yr + (size_t)slot_of[2 * t] * Dm;
    const __hip_bfloat16* r1 = Yr + (size_t)slot_of[2 * t + 1] * Dm;
    float sum = 0.f, sq = 0.f;
    for (int d = lane; d < Dm; d += 32) {
      float v = __bfloat162float(r0[d]) + __bfloat162float(r1[d]);
      yb[w * 770 + d] = v;
      sum += v; sq += v * v;
    }
#pragma unroll
    for (int off = 16; off; off >>= 1) {
      sum += __shfl_xor(sum, off, 32);
      sq  += __shfl_xor(sq,  off, 32);
    }
    if (lane == 0) {
      float mu = sum * (1.f / Dm);
      float var = sq * (1.f / Dm) - mu * mu;
      muS[w] = mu;
      rsS[w] = rsqrtf(var + LN_EPS);
    }
  }
  __syncthreads();
  int w = tid & 31, rr = tid >> 5;
  size_t basex = (size_t)b * Dm * 1024 + (size_t)h * 32;
  for (int it = 0; it < Dm / 8; ++it) {
    int d = it * 8 + rr;
    size_t gi = basex + (size_t)d * 1024 + w;
    out[gi] = (yb[w * 770 + d] - muS[w]) * rsS[w] * gamma[d] + beta[d] + x[gi];
  }
}

extern "C" void kernel_launch(void* const* d_in, const int* in_sizes, int n_in,
                              void* d_out, int out_size, void* d_ws, size_t ws_size,
                              hipStream_t stream) {
  (void)in_sizes; (void)n_in; (void)out_size; (void)ws_size;
  const float* x     = (const float*)d_in[0];
  const float* Wr    = (const float*)d_in[1];
  const float* W1    = (const float*)d_in[2];
  const float* W3    = (const float*)d_in[3];
  const float* W2    = (const float*)d_in[4];
  const float* gamma = (const float*)d_in[5];
  const float* beta  = (const float*)d_in[6];
  float* out = (float*)d_out;

  char* ws = (char*)d_ws;
  size_t o = 0;
  auto alloc = [&](size_t bytes) {
    void* p = ws + o;
    o += (bytes + 255) & ~(size_t)255;
    return p;
  };
  __hip_bfloat16* tokA = (__hip_bfloat16*)alloc((size_t)Tm * Dm * 2);
  __hip_bfloat16* W1t  = (__hip_bfloat16*)alloc((size_t)Em * EDm * Dm * 2);
  __hip_bfloat16* W3t  = (__hip_bfloat16*)alloc((size_t)Em * EDm * Dm * 2);
  __hip_bfloat16* W2t  = (__hip_bfloat16*)alloc((size_t)Em * Dm * EDm * 2);
  int*   eid    = (int*)  alloc((size_t)2 * Tm * 4);
  float* gate   = (float*)alloc((size_t)2 * Tm * 4);
  int*   counts = (int*)  alloc(256);
  int*   cursor = (int*)  alloc(256);
  int*   offs   = (int*)  alloc(256);
  int*   perm   = (int*)  alloc((size_t)CAPm * 4);
  float* gatep  = (float*)alloc((size_t)CAPm * 4);
  int*   slot_of= (int*)  alloc((size_t)2 * Tm * 4);
  __hip_bfloat16* Hbuf = (__hip_bfloat16*)alloc((size_t)CAPm * EDm * 2);
  __hip_bfloat16* Yr   = (__hip_bfloat16*)alloc((size_t)CAPm * Dm * 2);

  k_init<<<(CAPm + 255) / 256, 256, 0, stream>>>(perm, gatep, counts, cursor);
  // x [32][768][1024] -> tokens bf16 [32][1024][768]
  k_transpose<<<dim3(1024 / 32, 768 / 32, 32), 256, 0, stream>>>(x, tokA, 768, 1024);
  // W1/W3 [e][768][192] -> [e][192][768] (K contiguous)
  k_transpose<<<dim3(192 / 32, 768 / 32, Em), 256, 0, stream>>>(W1, W1t, 768, 192);
  k_transpose<<<dim3(192 / 32, 768 / 32, Em), 256, 0, stream>>>(W3, W3t, 768, 192);
  // W2 [e][192][768] -> [e][768][192] (K contiguous)
  k_transpose<<<dim3(768 / 32, 192 / 32, Em), 256, 0, stream>>>(W2, W2t, 192, 768);
  k_router<<<Tm / 8, 256, 0, stream>>>(tokA, Wr, eid, gate, counts);
  k_scan<<<1, 32, 0, stream>>>(counts, offs);
  k_scatter<<<(2 * Tm) / 256, 256, 0, stream>>>(eid, gate, offs, cursor, perm, gatep, slot_of);
  k_gemm1<<<CAPm / 64, 256, 0, stream>>>(tokA, W1t, W3t, offs, perm, Hbuf);
  k_gemm2<<<dim3(CAPm / 64, 2), 256, 0, stream>>>(Hbuf, W2t, offs, gatep, Yr);
  k_final<<<32 * 32, 256, 0, stream>>>(Yr, slot_of, x, gamma, beta, out);
}
